// DecoupledPointJAFAR_41704132444577
// MI455X (gfx1250) — matrix operand order repaired and checked by reading, not compiled, not run on hardware
//
#include <hip/hip_runtime.h>
#include <math.h>

// ---------------------------------------------------------------------------
// DecoupledPointJAFAR forward for MI455X (gfx1250, wave32, WMMA).
//
// Shapes: B=2, N=16384 (hr), M=8192 (lr), C=64, KNN=16.
// Heavy parts: kNN scan (268M candidate dists, selection-bound -> scalar
// insertion top-16 in registers) and the 64x64 pointwise layers
// (ge2/Q/K/pe -> v_wmma_f32_16x16x32_bf16, bf16 in / f32 accumulate).
// BatchNorm (training mode) via deterministic two-pass stats; rel-pos BN via
// analytic 3x3 covariance (avoids a 134MB intermediate). No atomics.
// ---------------------------------------------------------------------------

#define BB  2
#define NN  16384
#define MM  8192
#define KNB 16
#define EPSN 1e-5f

typedef __attribute__((ext_vector_type(16))) __bf16 v16bf;
typedef __attribute__((ext_vector_type(8)))  float  v8f;

// ---------------- pointwise 18 -> 64 (feat = concat(geo, rgb)) -------------
__global__ void __launch_bounds__(256) k_pw18(
    const float* __restrict__ geo, const float* __restrict__ rgb,
    const float* __restrict__ W, const float* __restrict__ bias,
    float* __restrict__ out, int P) {
  __shared__ float w[64 * 18];
  __shared__ float bsh[64];
  for (int i = threadIdx.x; i < 64 * 18; i += 256) w[i] = W[i];
  for (int i = threadIdx.x; i < 64; i += 256) bsh[i] = bias[i];
  __syncthreads();
  int gid = blockIdx.x * 256 + threadIdx.x;
  if (gid >= BB * P) return;
  int b = gid / P, p = gid % P;
  float f[18];
#pragma unroll
  for (int j = 0; j < 9; ++j) f[j] = geo[((size_t)b * 9 + j) * P + p];
#pragma unroll
  for (int j = 0; j < 9; ++j) f[9 + j] = rgb[((size_t)b * 9 + j) * P + p];
  for (int c = 0; c < 64; ++c) {
    float s = bsh[c];
#pragma unroll
    for (int j = 0; j < 18; ++j) s += w[c * 18 + j] * f[j];
    out[((size_t)b * 64 + c) * P + p] = s;
  }
}

// ------------- per-channel BN stats (one block per channel) ----------------
__global__ void __launch_bounds__(256) k_chan_stats(
    const float* __restrict__ X, int P, int Cn,
    const float* __restrict__ g, const float* __restrict__ bt,
    float2* __restrict__ bn) {
  int c = blockIdx.x;
  __shared__ float s1[256], s2[256];
  float a1 = 0.f, a2 = 0.f;
  for (int i = threadIdx.x; i < BB * P; i += 256) {
    int b = i / P, p = i % P;
    float x = X[((size_t)b * Cn + c) * P + p];
    a1 += x; a2 += x * x;
  }
  s1[threadIdx.x] = a1; s2[threadIdx.x] = a2;
  __syncthreads();
  for (int s = 128; s > 0; s >>= 1) {
    if (threadIdx.x < s) {
      s1[threadIdx.x] += s1[threadIdx.x + s];
      s2[threadIdx.x] += s2[threadIdx.x + s];
    }
    __syncthreads();
  }
  if (threadIdx.x == 0) {
    float inv = 1.0f / (float)(BB * P);
    float m = s1[0] * inv;
    float v = s2[0] * inv - m * m;
    float sc = rsqrtf(v + EPSN) * g[c];
    bn[c] = make_float2(sc, bt[c] - m * sc);
  }
}

// -------- generic pointwise 64 -> OUT via WMMA bf16 (BN+ReLU on input) -----
// Y[b][o][p] = bias[o] + sum_c W[o][c] * act(X[b][c][p])
__global__ void __launch_bounds__(256) k_pw64(
    const float* __restrict__ X, float* __restrict__ Y,
    const float* __restrict__ W, const float* __restrict__ bias,
    const float2* __restrict__ bnin, int relu_in, int OUT, int P) {
  __shared__ __bf16 wlds[64][64];
  __shared__ __bf16 xlds[64][128];
  int b = blockIdx.y;
  int pbase = blockIdx.x * 128;
  for (int i = threadIdx.x; i < OUT * 64; i += 256)
    wlds[i >> 6][i & 63] = (__bf16)W[i];
  for (int i = threadIdx.x; i < 64 * 128; i += 256) {
    int cin = i >> 7, p = i & 127;
    float x = X[((size_t)b * 64 + cin) * P + pbase + p];
    if (bnin) {
      float2 sb = bnin[cin];
      x = x * sb.x + sb.y;
      if (relu_in) x = fmaxf(x, 0.f);
    }
    xlds[cin][p] = (__bf16)x;
  }
  __syncthreads();
  int lane = threadIdx.x & 31;
  int wave = threadIdx.x >> 5;
  int col  = lane & 15;        // point within tile / A row within tile
  int hi   = lane >> 4;        // half-wave selector
  int pt   = wave * 16;        // this wave's 16-point slice
  int ntiles = OUT >> 4;
  for (int tt = 0; tt < ntiles; ++tt) {
    v8f acc = {};
#pragma unroll
    for (int ks = 0; ks < 64; ks += 32) {
      v16bf a, bm;
#pragma unroll
      for (int i = 0; i < 8; ++i) {
        a[i]     = wlds[tt * 16 + col][ks + hi * 8 + i];
        a[8 + i] = wlds[tt * 16 + col][ks + 16 + hi * 8 + i];
      }
#pragma unroll
      for (int i = 0; i < 16; ++i)
        bm[i] = xlds[ks + (i < 8 ? hi * 8 + i : 16 + hi * 8 + (i - 8))][pt + col];
      acc = __builtin_amdgcn_wmma_f32_16x16x32_bf16(
          false, a, false, bm, (short)0, acc, false, false);
    }
#pragma unroll
    for (int r = 0; r < 8; ++r) {
      int row = tt * 16 + r + hi * 8;
      Y[((size_t)b * OUT + row) * P + pbase + pt + col] = acc[r] + bias[row];
    }
  }
}

// ---------------- geom = relu(bn(pre2)) (hr path) --------------------------
__global__ void __launch_bounds__(256) k_geom_fin(
    const float* __restrict__ pre2, const float2* __restrict__ bn,
    float* __restrict__ out, int P) {
  size_t i = (size_t)blockIdx.x * 256 + threadIdx.x;
  if (i >= (size_t)BB * 64 * P) return;
  int c = (int)((i / P) % 64);
  float2 sb = bn[c];
  out[i] = fmaxf(pre2[i] * sb.x + sb.y, 0.f);
}

// ------- geom_lr = relu(bn(pre2)) * (scale+1) + shift (FiLM, lr path) ------
__global__ void __launch_bounds__(256) k_geom_film(
    const float* __restrict__ pre2, const float2* __restrict__ bn,
    const float* __restrict__ val,
    const float* __restrict__ scw, const float* __restrict__ scb,
    const float* __restrict__ shw, const float* __restrict__ shb,
    float* __restrict__ out, int P) {
  __shared__ float sw[64 * 6], hw[64 * 6], sb[64], hb[64];
  for (int i = threadIdx.x; i < 64 * 6; i += 256) { sw[i] = scw[i]; hw[i] = shw[i]; }
  for (int i = threadIdx.x; i < 64; i += 256) { sb[i] = scb[i]; hb[i] = shb[i]; }
  __syncthreads();
  int gid = blockIdx.x * 256 + threadIdx.x;
  if (gid >= BB * P) return;
  int b = gid / P, p = gid % P;
  float v[6];
#pragma unroll
  for (int j = 0; j < 6; ++j) v[j] = val[((size_t)b * 6 + j) * P + p];
  for (int c = 0; c < 64; ++c) {
    float2 nb = bn[c];
    float gm = fmaxf(pre2[((size_t)b * 64 + c) * P + p] * nb.x + nb.y, 0.f);
    float sc = sb[c], sh = hb[c];
#pragma unroll
    for (int j = 0; j < 6; ++j) { sc += sw[c * 6 + j] * v[j]; sh += hw[c * 6 + j] * v[j]; }
    out[((size_t)b * 64 + c) * P + p] = gm * (sc + 1.f) + sh;
  }
}

// --------------- boundary head: sigmoid(w2 . relu(bn(bd_pre))) -------------
__global__ void __launch_bounds__(256) k_bdy(
    const float* __restrict__ bdpre, const float2* __restrict__ bn,
    const float* __restrict__ w2, const float* __restrict__ b2,
    float* __restrict__ out, int P) {
  int gid = blockIdx.x * 256 + threadIdx.x;
  if (gid >= BB * P) return;
  int b = gid / P, p = gid % P;
  float s = b2[0];
#pragma unroll
  for (int j = 0; j < 32; ++j) {
    float2 nb = bn[j];
    float h = fmaxf(bdpre[((size_t)b * 32 + j) * P + p] * nb.x + nb.y, 0.f);
    s += w2[j] * h;
  }
  out[(size_t)b * P + p] = 1.f / (1.f + __expf(-s));
}

// ---------------------- kNN: top-16 by insertion ---------------------------
__global__ void __launch_bounds__(256) k_knn(
    const float* __restrict__ xh, const float* __restrict__ xl,
    int* __restrict__ kidx) {
  __shared__ float lx[3][256];
  int gid = blockIdx.x * 256 + threadIdx.x;
  int b = gid / NN, n = gid % NN;            // uniform b per block (NN%256==0)
  float ax = xh[((size_t)b * 3 + 0) * NN + n];
  float ay = xh[((size_t)b * 3 + 1) * NN + n];
  float az = xh[((size_t)b * 3 + 2) * NN + n];
  float best[KNB]; int bid[KNB];
#pragma unroll
  for (int i = 0; i < KNB; ++i) { best[i] = 3.4e38f; bid[i] = 0; }
  for (int tb = 0; tb < MM; tb += 256) {
    __syncthreads();
    for (int i = threadIdx.x; i < 768; i += 256)
      lx[i >> 8][i & 255] = xl[((size_t)b * 3 + (i >> 8)) * MM + tb + (i & 255)];
    __syncthreads();
    for (int j = 0; j < 256; ++j) {
      float dx = ax - lx[0][j], dy = ay - lx[1][j], dz = az - lx[2][j];
      float d2 = dx * dx + dy * dy + dz * dz;
      if (d2 < best[KNB - 1]) {
        float cd = d2; int ci = tb + j;
#pragma unroll
        for (int s = 0; s < KNB; ++s) {
          if (cd < best[s]) {
            float td = best[s]; int ti = bid[s];
            best[s] = cd; bid[s] = ci; cd = td; ci = ti;
          }
        }
      }
    }
  }
#pragma unroll
  for (int i = 0; i < KNB; ++i)
    kidx[((size_t)b * NN + n) * KNB + i] = bid[i];
}

// ---- rel_pos first/second moments (9 sums) for analytic BN on rp layer1 ---
__global__ void __launch_bounds__(256) k_relstats(
    const float* __restrict__ xh, const float* __restrict__ xl,
    const int* __restrict__ kidx, float* __restrict__ partial) {
  float acc[9];
#pragma unroll
  for (int q = 0; q < 9; ++q) acc[q] = 0.f;
  for (int g = blockIdx.x * 256 + threadIdx.x; g < BB * NN; g += gridDim.x * 256) {
    int b = g / NN, n = g % NN;
    float ax = xh[((size_t)b * 3 + 0) * NN + n];
    float ay = xh[((size_t)b * 3 + 1) * NN + n];
    float az = xh[((size_t)b * 3 + 2) * NN + n];
#pragma unroll
    for (int k = 0; k < KNB; ++k) {
      int id = kidx[((size_t)b * NN + n) * KNB + k];
      float rx = ax - xl[((size_t)b * 3 + 0) * MM + id];
      float ry = ay - xl[((size_t)b * 3 + 1) * MM + id];
      float rz = az - xl[((size_t)b * 3 + 2) * MM + id];
      acc[0] += rx; acc[1] += ry; acc[2] += rz;
      acc[3] += rx * rx; acc[4] += rx * ry; acc[5] += rx * rz;
      acc[6] += ry * ry; acc[7] += ry * rz; acc[8] += rz * rz;
    }
  }
  __shared__ float red[256];
  for (int q = 0; q < 9; ++q) {
    red[threadIdx.x] = acc[q];
    __syncthreads();
    for (int s = 128; s > 0; s >>= 1) {
      if (threadIdx.x < s) red[threadIdx.x] += red[threadIdx.x + s];
      __syncthreads();
    }
    if (threadIdx.x == 0) partial[blockIdx.x * 9 + q] = red[0];
    __syncthreads();
  }
}

__global__ void k_relred(
    const float* __restrict__ partial, int nblk,
    const float* __restrict__ w1, const float* __restrict__ b1,
    const float* __restrict__ g, const float* __restrict__ bt,
    float2* __restrict__ bnrp) {
  __shared__ float S[9];
  if (threadIdx.x < 9) {
    float s = 0.f;
    for (int i = 0; i < nblk; ++i) s += partial[i * 9 + threadIdx.x];
    S[threadIdx.x] = s;
  }
  __syncthreads();
  int c = threadIdx.x;
  if (c < 64) {
    const float inv = 1.0f / (float)((size_t)BB * NN * KNB);
    float mx = S[0] * inv, my = S[1] * inv, mz = S[2] * inv;
    float cxx = S[3] * inv - mx * mx, cxy = S[4] * inv - mx * my;
    float cxz = S[5] * inv - mx * mz, cyy = S[6] * inv - my * my;
    float cyz = S[7] * inv - my * mz, czz = S[8] * inv - mz * mz;
    float wx = w1[c * 3], wy = w1[c * 3 + 1], wz = w1[c * 3 + 2];
    float mh = wx * mx + wy * my + wz * mz + b1[c];
    float var = wx * wx * cxx + wy * wy * cyy + wz * wz * czz
              + 2.f * (wx * wy * cxy + wx * wz * cxz + wy * wz * cyz);
    float sc = rsqrtf(var + EPSN) * g[c];
    bnrp[c] = make_float2(sc, bt[c] - mh * sc);
  }
}

// ---- fused: pe MLP (WMMA) + K gather + QK dot + softmax + value mix -------
#define AW 4
__global__ void __launch_bounds__(128) k_attn(
    const float* __restrict__ Q, const float* __restrict__ K,
    const float* __restrict__ val, const float* __restrict__ xh,
    const float* __restrict__ xl, const int* __restrict__ kidx,
    const float* __restrict__ w1, const float* __restrict__ b1,
    const float2* __restrict__ bnrp,
    const float* __restrict__ w2, const float* __restrict__ b2,
    float* __restrict__ rec) {
  __shared__ __bf16 w2s[64][64];
  __shared__ __bf16 hs[AW][64][KNB];
  __shared__ float  kg[AW][64][KNB];
  __shared__ float  qs[AW][64];
  __shared__ int    ids[AW][KNB];

  for (int i = threadIdx.x; i < 64 * 64; i += 128)
    w2s[i >> 6][i & 63] = (__bf16)w2[i];

  int wave = threadIdx.x >> 5, lane = threadIdx.x & 31;
  int ptid = blockIdx.x * AW + wave;            // grid is exact: EXEC all-ones
  int b = ptid / NN, n = ptid % NN;
  if (lane < KNB) ids[wave][lane] = kidx[(size_t)ptid * KNB + lane];
  qs[wave][lane]      = Q[((size_t)b * 64 + lane) * NN + n];
  qs[wave][lane + 32] = Q[((size_t)b * 64 + lane + 32) * NN + n];
  __syncthreads();

  int nb = lane & 15, hi = lane >> 4;
  float ax = xh[((size_t)b * 3 + 0) * NN + n];
  float ay = xh[((size_t)b * 3 + 1) * NN + n];
  float az = xh[((size_t)b * 3 + 2) * NN + n];
  int id = ids[wave][nb];
  float rx = ax - xl[((size_t)b * 3 + 0) * MM + id];
  float ry = ay - xl[((size_t)b * 3 + 1) * MM + id];
  float rz = az - xl[((size_t)b * 3 + 2) * MM + id];
#pragma unroll
  for (int i = 0; i < 32; ++i) {
    int ch = 2 * i + hi;                        // evens / odds split per half
    float t = w1[ch * 3] * rx + w1[ch * 3 + 1] * ry + w1[ch * 3 + 2] * rz + b1[ch];
    float2 sb = bnrp[ch];
    t = fmaxf(t * sb.x + sb.y, 0.f);
    hs[wave][ch][nb] = (__bf16)t;
    kg[wave][ch][nb] = K[((size_t)b * 64 + ch) * MM + id];
  }
  __syncthreads();

  float part = 0.f;
#pragma unroll
  for (int tt = 0; tt < 4; ++tt) {
    v8f acc = {};
#pragma unroll
    for (int ks = 0; ks < 64; ks += 32) {
      v16bf a, bm;
#pragma unroll
      for (int i = 0; i < 8; ++i) {
        a[i]     = w2s[tt * 16 + nb][ks + hi * 8 + i];
        a[8 + i] = w2s[tt * 16 + nb][ks + 16 + hi * 8 + i];
      }
#pragma unroll
      for (int i = 0; i < 16; ++i)
        bm[i] = hs[wave][ks + (i < 8 ? hi * 8 + i : 16 + hi * 8 + (i - 8))][nb];
      acc = __builtin_amdgcn_wmma_f32_16x16x32_bf16(
          false, a, false, bm, (short)0, acc, false, false);
    }
#pragma unroll
    for (int r = 0; r < 8; ++r) {
      int ch = tt * 16 + r + hi * 8;
      float pe = acc[r] + b2[ch];
      part += qs[wave][ch] * (pe + kg[wave][ch][nb]);
    }
  }
  // combine the two half-wave partials for each neighbor column
  float logit = (part + __shfl_xor(part, 16, 32)) * 0.125f;   // /sqrt(64)
  float mx = logit;
#pragma unroll
  for (int d = 8; d >= 1; d >>= 1) mx = fmaxf(mx, __shfl_xor(mx, d, 32));
  float e = __expf(logit - mx);
  float se = e;
#pragma unroll
  for (int d = 8; d >= 1; d >>= 1) se += __shfl_xor(se, d, 32);
  float attn = e / se;
#pragma unroll
  for (int c = 0; c < 6; ++c) {
    float v = val[((size_t)b * 6 + c) * MM + id] * attn;
#pragma unroll
    for (int d = 8; d >= 1; d >>= 1) v += __shfl_xor(v, d, 32);
    if (lane == 0) rec[((size_t)b * 6 + c) * NN + n] = v;
  }
}

// ---------------------------------------------------------------------------
extern "C" void kernel_launch(void* const* d_in, const int* in_sizes, int n_in,
                              void* d_out, int out_size, void* d_ws, size_t ws_size,
                              hipStream_t stream) {
  (void)in_sizes; (void)n_in; (void)out_size; (void)ws_size;
  const float* xyz_hr = (const float*)d_in[0];
  const float* xyz_lr = (const float*)d_in[1];
  const float* val_lr = (const float*)d_in[2];
  const float* geo_hr = (const float*)d_in[3];
  const float* geo_lr = (const float*)d_in[4];
  const float* rgb_hr = (const float*)d_in[5];
  const float* rgb_lr = (const float*)d_in[6];
  const float* ge_w1 = (const float*)d_in[7];
  const float* ge_b1 = (const float*)d_in[8];
  const float* ge_g1 = (const float*)d_in[9];
  const float* ge_bt1 = (const float*)d_in[10];
  const float* ge_w2 = (const float*)d_in[11];
  const float* ge_b2 = (const float*)d_in[12];
  const float* ge_g2 = (const float*)d_in[13];
  const float* ge_bt2 = (const float*)d_in[14];
  const float* sc_w = (const float*)d_in[15];
  const float* sc_b = (const float*)d_in[16];
  const float* sh_w = (const float*)d_in[17];
  const float* sh_b = (const float*)d_in[18];
  const float* q_w = (const float*)d_in[19];
  const float* q_b = (const float*)d_in[20];
  const float* k_w = (const float*)d_in[21];
  const float* k_b = (const float*)d_in[22];
  const float* bd_w1 = (const float*)d_in[23];
  const float* bd_b1 = (const float*)d_in[24];
  const float* bd_g = (const float*)d_in[25];
  const float* bd_bt = (const float*)d_in[26];
  const float* bd_w2 = (const float*)d_in[27];
  const float* bd_b2 = (const float*)d_in[28];
  const float* rp_w1 = (const float*)d_in[29];
  const float* rp_b1 = (const float*)d_in[30];
  const float* rp_g = (const float*)d_in[31];
  const float* rp_bt = (const float*)d_in[32];
  const float* rp_w2 = (const float*)d_in[33];
  const float* rp_b2 = (const float*)d_in[34];

  float* out_rec = (float*)d_out;                       // [2,6,16384]
  float* out_bdy = (float*)d_out + (size_t)BB * 6 * NN; // [2,1,16384]

  // workspace carve-up (buffer reuse: pre1->geom, pre2->Q/K)
  char* ws = (char*)d_ws;
  size_t off = 0;
  auto carve = [&](size_t bytes) {
    off = (off + 255) & ~(size_t)255;
    void* p = ws + off;
    off += bytes;
    return p;
  };
  float* bufA_hr = (float*)carve((size_t)BB * 64 * NN * 4); // pre1_hr / geom_hr
  float* bufA_lr = (float*)carve((size_t)BB * 64 * MM * 4); // pre1_lr / geom_lr
  float* bufB_hr = (float*)carve((size_t)BB * 64 * NN * 4); // pre2_hr / Q
  float* bufB_lr = (float*)carve((size_t)BB * 64 * MM * 4); // pre2_lr / K
  float* bd_pre  = (float*)carve((size_t)BB * 32 * NN * 4);
  int*   kidx    = (int*)  carve((size_t)BB * NN * KNB * 4);
  float2* bn1h = (float2*)carve(64 * 8);
  float2* bn1l = (float2*)carve(64 * 8);
  float2* bn2h = (float2*)carve(64 * 8);
  float2* bn2l = (float2*)carve(64 * 8);
  float2* bnbd = (float2*)carve(32 * 8);
  float2* bnrp = (float2*)carve(64 * 8);
  float* relpart = (float*)carve(64 * 9 * 4);

  // 1) feat -> pre1 (18->64)
  k_pw18<<<dim3((BB * NN + 255) / 256), 256, 0, stream>>>(geo_hr, rgb_hr, ge_w1, ge_b1, bufA_hr, NN);
  k_pw18<<<dim3((BB * MM + 255) / 256), 256, 0, stream>>>(geo_lr, rgb_lr, ge_w1, ge_b1, bufA_lr, MM);
  // 2) BN1 stats (separate stats for hr/lr, as in reference)
  k_chan_stats<<<64, 256, 0, stream>>>(bufA_hr, NN, 64, ge_g1, ge_bt1, bn1h);
  k_chan_stats<<<64, 256, 0, stream>>>(bufA_lr, MM, 64, ge_g1, ge_bt1, bn1l);
  // 3) pre2 = ge_w2 * relu(bn1(pre1)) + b2   [WMMA]
  k_pw64<<<dim3(NN / 128, BB), 256, 0, stream>>>(bufA_hr, bufB_hr, ge_w2, ge_b2, bn1h, 1, 64, NN);
  k_pw64<<<dim3(MM / 128, BB), 256, 0, stream>>>(bufA_lr, bufB_lr, ge_w2, ge_b2, bn1l, 1, 64, MM);
  // 4) BN2 stats
  k_chan_stats<<<64, 256, 0, stream>>>(bufB_hr, NN, 64, ge_g2, ge_bt2, bn2h);
  k_chan_stats<<<64, 256, 0, stream>>>(bufB_lr, MM, 64, ge_g2, ge_bt2, bn2l);
  // 5) geom_hr = relu(bn2); geom_lr = relu(bn2) FiLM-modulated by val_lr
  k_geom_fin<<<dim3(((size_t)BB * 64 * NN + 255) / 256), 256, 0, stream>>>(bufB_hr, bn2h, bufA_hr, NN);
  k_geom_film<<<dim3((BB * MM + 255) / 256), 256, 0, stream>>>(bufB_lr, bn2l, val_lr, sc_w, sc_b, sh_w, sh_b, bufA_lr, MM);
  // 6) Q, K, boundary projection  [WMMA]
  k_pw64<<<dim3(NN / 128, BB), 256, 0, stream>>>(bufA_hr, bufB_hr, q_w, q_b, nullptr, 0, 64, NN);
  k_pw64<<<dim3(MM / 128, BB), 256, 0, stream>>>(bufA_lr, bufB_lr, k_w, k_b, nullptr, 0, 64, MM);
  k_pw64<<<dim3(NN / 128, BB), 256, 0, stream>>>(bufA_hr, bd_pre, bd_w1, bd_b1, nullptr, 0, 32, NN);
  // 7) boundary BN stats + head
  k_chan_stats<<<32, 256, 0, stream>>>(bd_pre, NN, 32, bd_g, bd_bt, bnbd);
  k_bdy<<<dim3((BB * NN + 255) / 256), 256, 0, stream>>>(bd_pre, bnbd, bd_w2, bd_b2, out_bdy, NN);
  // 8) kNN
  k_knn<<<dim3(BB * NN / 256), 256, 0, stream>>>(xyz_hr, xyz_lr, kidx);
  // 9) analytic BN for rel-pos layer1
  k_relstats<<<64, 256, 0, stream>>>(xyz_hr, xyz_lr, kidx, relpart);
  k_relred<<<1, 64, 0, stream>>>(relpart, 64, rp_w1, rp_b1, rp_g, rp_bt, bnrp);
  // 10) fused pe-MLP + attention + value mix  [WMMA]
  k_attn<<<dim3(BB * NN / AW), 128, 0, stream>>>(bufB_hr, bufB_lr, val_lr, xyz_hr, xyz_lr,
                                                 kidx, rp_w1, rp_b1, bnrp, rp_w2, rp_b2, out_rec);
}
